// Decoder_55164559950059
// MI455X (gfx1250) — compile-verified, matrix operand order checked by reading, not run on hardware
//
#include <hip/hip_runtime.h>
#include <hip/hip_bf16.h>

typedef __attribute__((ext_vector_type(16))) __bf16 v16bf;
typedef __attribute__((ext_vector_type(8)))  float  v8f;

#define NB   8
#define NQ   500
#define NC   256
#define NS   8
#define NROWS (NB*NQ*NS)          // 32000

// ---------------- workspace layout (bytes) ----------------
// emb bf16 : 32000*256*2 = 16,384,000   @ 0
// W1T bf16 : 256*256*2   =    131,072   @ 16,384,000
// sp  f32  : 32000*2*4   =    256,000   @ 16,515,072
#define WS_EMB 0
#define WS_W1T 16384000
#define WS_SP  16515072

// ---------------------------------------------------------------------------
// Stage 1: polynomial eval + per-level bilinear sampling -> emb (bf16 rows)
// One block per (b,q,s) row, one thread per channel. Each neighbor fetch is a
// contiguous, fully coalesced 1KB read (memory layout is (B, 8500, C)).
// ---------------------------------------------------------------------------
__global__ void sample_kernel(const float* __restrict__ ref_polys,
                              const int*   __restrict__ ref_levels,
                              const float* __restrict__ memory,
                              __bf16*      __restrict__ embbf,
                              float*       __restrict__ sp)
{
    const int r = blockIdx.x;           // row in [0, 32000)
    const int c = threadIdx.x;          // channel
    const int s = r & 7;
    const int q = (r >> 3) % NQ;
    const int b = r / (NQ * NS);

    const float lam = (float)s * (1.0f / 7.0f);
    const float l2 = lam * lam, l3 = l2 * lam;
    const float* rp = ref_polys + (size_t)(b * NQ + q) * 8;
    const float px = rp[0]*l3 + rp[1]*l2 + rp[2]*lam + rp[3];
    const float py = rp[4]*l3 + rp[5]*l2 + rp[6]*lam + rp[7];
    const float sx = 2.0f * (px - 0.5f);
    const float sy = 2.0f * (py - 0.5f);

    const int lvl = ref_levels[b * NQ + q];
    const int hw_tab[4] = {80, 40, 20, 10};
    const int st_tab[4] = {0, 6400, 8000, 8400};
    const int w  = hw_tab[lvl];
    const int h  = hw_tab[lvl];
    const int st = st_tab[lvl];

    const float gx = (sx + 1.0f) * 0.5f * (float)w - 0.5f;
    const float gy = (sy + 1.0f) * 0.5f * (float)h - 0.5f;
    const float x0f = floorf(gx), y0f = floorf(gy);
    const int   x0 = (int)x0f,    y0 = (int)y0f;
    const float wx1 = gx - x0f, wx0 = 1.0f - wx1;
    const float wy1 = gy - y0f, wy0 = 1.0f - wy1;

    const float* mb = memory + ((size_t)b * 8500 + st) * NC;

    const int   xs[2] = {x0, x0 + 1}, ys[2] = {y0, y0 + 1};
    const float wxs[2] = {wx0, wx1},  wys[2] = {wy0, wy1};

    float acc = 0.0f;
#pragma unroll
    for (int iy = 0; iy < 2; ++iy) {
#pragma unroll
        for (int ix = 0; ix < 2; ++ix) {
            const int xi = xs[ix], yi = ys[iy];
            const bool valid = (xi >= 0) && (xi < w) && (yi >= 0) && (yi < h);
            const int xc = min(max(xi, 0), w - 1);
            const int yc = min(max(yi, 0), h - 1);
            const float v = mb[(size_t)(yc * w + xc) * NC + c];
            acc += (valid ? wys[iy] * wxs[ix] : 0.0f) * v;
        }
    }

    embbf[(size_t)r * NC + c] = (__bf16)acc;
    if (c == 0) { sp[r * 2 + 0] = sx; sp[r * 2 + 1] = sy; }
}

// ---------------------------------------------------------------------------
// Stage 2: W1 -> W1^T in bf16 (one-time, 128KB) so that WMMA B-fragments are
// two contiguous 16-byte loads per lane.
// ---------------------------------------------------------------------------
__global__ void w1t_kernel(const float* __restrict__ W1, __bf16* __restrict__ W1T)
{
    const int k = threadIdx.x;          // input channel (coalesced write)
    const int n = blockIdx.x;           // output channel
    W1T[(size_t)n * NC + k] = (__bf16)W1[(size_t)k * NC + n];
}

// ---------------------------------------------------------------------------
// Stage 3: fused MLP. 16-row M-tile per block; 4 waves x 4 N-tiles = 256 cols.
// Layer 1 via v_wmma_f32_16x16x32_bf16 (8 K-steps), bias+tanh in regs,
// h staged to LDS (stride 260 = conflict-free), layer 2 (256->4) via VALU,
// final tanh/scale/add + scatter to output layout (B,Q,S*NP,2).
// ---------------------------------------------------------------------------
__global__ void __launch_bounds__(128)
mlp_kernel(const __bf16* __restrict__ embbf,
           const __bf16* __restrict__ W1T,
           const float*  __restrict__ sp,
           const float*  __restrict__ b1,
           const float*  __restrict__ W2,
           const float*  __restrict__ b2,
           float*        __restrict__ out)
{
    __shared__ float hsm[16 * 260];

    const int t    = threadIdx.x;
    const int wave = t >> 5;
    const int lane = t & 31;
    const int g    = lane >> 4;         // lane group (0/1)
    const int nl   = lane & 15;
    const int mbase = blockIdx.x * 16;

    union FragU { uint4 qp[2]; v16bf v; };

    v8f acc[4] = {v8f{0.f,0.f,0.f,0.f,0.f,0.f,0.f,0.f},
                  v8f{0.f,0.f,0.f,0.f,0.f,0.f,0.f,0.f},
                  v8f{0.f,0.f,0.f,0.f,0.f,0.f,0.f,0.f},
                  v8f{0.f,0.f,0.f,0.f,0.f,0.f,0.f,0.f}};

    for (int kb = 0; kb < NC; kb += 32) {
        // A fragment: 16-bit A 16x32 layout — lane m=nl, lanes 16-31 take the
        // K+8 / K+24 halves; both chunks are 16B-aligned contiguous loads.
        FragU a;
        const __bf16* arow = embbf + (size_t)(mbase + nl) * NC + kb + g * 8;
        a.qp[0] = *reinterpret_cast<const uint4*>(arow);
        a.qp[1] = *reinterpret_cast<const uint4*>(arow + 16);

#pragma unroll
        for (int nt = 0; nt < 4; ++nt) {
            const int ncol = wave * 64 + nt * 16 + nl;
            FragU bf;
            const __bf16* brow = W1T + (size_t)ncol * NC + kb + g * 8;
            bf.qp[0] = *reinterpret_cast<const uint4*>(brow);
            bf.qp[1] = *reinterpret_cast<const uint4*>(brow + 16);
            acc[nt] = __builtin_amdgcn_wmma_f32_16x16x32_bf16(
                false, a.v, false, bf.v, (short)0, acc[nt], false, false);
        }
    }

    // C/D layout: element e of acc -> row = e + 8*g, col = nl (+ tile offset)
#pragma unroll
    for (int nt = 0; nt < 4; ++nt) {
        const int ncol = wave * 64 + nt * 16 + nl;
        const float bias = b1[ncol];
#pragma unroll
        for (int e = 0; e < 8; ++e) {
            const int row = e + 8 * g;
            hsm[row * 260 + ncol] = tanhf(acc[nt][e] + bias);
        }
    }
    __syncthreads();

    if (t < 64) {                       // 16 rows x 4 outputs
        const int row = t >> 2;
        const int j   = t & 3;          // j = np*2 + xy
        float o = b2[j];
        for (int cc = 0; cc < NC; ++cc)
            o += hsm[row * 260 + cc] * W2[cc * 4 + j];

        const int r  = mbase + row;
        const int s  = r & 7;
        const int q  = (r >> 3) % NQ;
        const int b  = r / (NQ * NS);
        const int np = j >> 1;
        const int xy = j & 1;
        const float val = 0.077f * tanhf(o) + sp[r * 2 + xy];
        out[((size_t)(b * NQ + q) * 16 + (s * 2 + np)) * 2 + xy] = val;
    }
}

// ---------------------------------------------------------------------------
extern "C" void kernel_launch(void* const* d_in, const int* in_sizes, int n_in,
                              void* d_out, int out_size, void* d_ws, size_t ws_size,
                              hipStream_t stream)
{
    const float* ref_polys  = (const float*)d_in[0];
    const int*   ref_levels = (const int*)  d_in[1];
    const float* memory     = (const float*)d_in[2];
    const float* W1         = (const float*)d_in[3];
    const float* b1         = (const float*)d_in[4];
    const float* W2         = (const float*)d_in[5];
    const float* b2         = (const float*)d_in[6];

    char* ws = (char*)d_ws;
    __bf16* embbf = (__bf16*)(ws + WS_EMB);
    __bf16* W1T   = (__bf16*)(ws + WS_W1T);
    float*  sp    = (float*)(ws + WS_SP);
    float*  out   = (float*)d_out;

    sample_kernel<<<NROWS, NC, 0, stream>>>(ref_polys, ref_levels, memory, embbf, sp);
    w1t_kernel<<<NC, NC, 0, stream>>>(W1, W1T);
    mlp_kernel<<<NROWS / 16, 128, 0, stream>>>(embbf, W1T, sp, b1, W2, b2, out);
}
